// TensorParallelThinkerDecoderLayer_14233521619264
// MI455X (gfx1250) — compile-verified, matrix operand order checked by reading, not run on hardware
//
#include <hip/hip_runtime.h>
#include <hip/hip_bf16.h>

typedef __bf16 bf16_t;
typedef __attribute__((ext_vector_type(16))) __bf16 v16bf;
typedef __attribute__((ext_vector_type(8)))  float  v8f;
typedef __attribute__((ext_vector_type(4)))  int    v4i;

constexpr int B_ = 2, T_ = 1024, H_ = 2048, NH_ = 32, NKV_ = 8, D_ = 64;
constexpr int E_ = 16, KTOP_ = 2, I_ = 1024, SI_ = 1024;
constexpr int NT_ = B_ * T_;

#if defined(__has_builtin)
#if __has_builtin(__builtin_amdgcn_global_load_async_to_lds_b128)
#define HAS_ASYNC_LDS 1
#endif
#endif
#ifndef HAS_ASYNC_LDS
#define HAS_ASYNC_LDS 0
#endif

typedef __attribute__((address_space(1))) v4i gv4i_t;
typedef __attribute__((address_space(3))) v4i lv4i_t;

// 16-byte global -> LDS copy through the CDNA5 async engine (ASYNCcnt),
// with a synchronous fallback if the builtin is unavailable.
__device__ inline void async_copy16(const bf16_t* __restrict__ g, bf16_t* l) {
#if HAS_ASYNC_LDS
    __builtin_amdgcn_global_load_async_to_lds_b128((gv4i_t*)g, (lv4i_t*)l, 0, 0);
#else
    *(uint4*)l = *(const uint4*)g;
#endif
}
__device__ inline void async_join() {
#if HAS_ASYNC_LDS
    asm volatile("s_wait_asynccnt 0" ::: "memory");
#endif
}

// ---------------- WMMA fragment helpers (16x16x32 bf16) ----------------
// A-layout (16x32, M x K): lane = lo + 16*hi, row M = lo,
// element p -> K = (p/8)*16 + hi*8 + (p%8)
__device__ inline v16bf load_a_row(const bf16_t* __restrict__ row, int k0, int hi) {
    const bf16_t* p = row + k0 + hi * 8;
    v16bf a;
#pragma unroll
    for (int j = 0; j < 8; ++j) { a[j] = p[j]; a[8 + j] = p[16 + j]; }
    return a;
}

__device__ inline v16bf load_a_frag(const bf16_t* __restrict__ A, int lda,
                                    int m0, int k0, int lane) {
    int lo = lane & 15, hi = lane >> 4;
    return load_a_row(A + (size_t)(m0 + lo) * lda, k0, hi);
}

// B-layout (32x16, K x N): lane lo holds column N = n0+lo, K striped like A.
// Memory is K-major (row = k), loads stride by ldb. Used on LDS tiles.
__device__ inline v16bf load_b_strided(const bf16_t* __restrict__ Bm, int ldb,
                                       int k0, int n0, int lane) {
    int lo = lane & 15, hi = lane >> 4;
    v16bf b;
#pragma unroll
    for (int p = 0; p < 16; ++p) {
        int k = ((p >> 3) << 4) + hi * 8 + (p & 7);
        b[p] = Bm[(size_t)(k0 + k) * ldb + n0 + lo];
    }
    return b;
}

__device__ inline v8f wmma_bf(v16bf a, v16bf b, v8f c) {
    return __builtin_amdgcn_wmma_f32_16x16x32_bf16(false, a, false, b, (short)0, c,
                                                   false, false);
}

__device__ inline float silu(float x) { return x / (1.f + __expf(-x)); }

// ---------------- elementwise / reduction kernels ----------------
__global__ void f2bf_kernel(const float* __restrict__ in, bf16_t* __restrict__ out,
                            size_t n) {
    size_t i = (size_t)blockIdx.x * blockDim.x + threadIdx.x;
    size_t stride = (size_t)gridDim.x * blockDim.x;
    for (; i < n; i += stride) out[i] = (bf16_t)in[i];
}

__global__ void zero_f_kernel(float* __restrict__ p, size_t n) {
    size_t i = (size_t)blockIdx.x * blockDim.x + threadIdx.x;
    size_t stride = (size_t)gridDim.x * blockDim.x;
    for (; i < n; i += stride) p[i] = 0.f;
}

__global__ void zero_i_kernel(int* __restrict__ p, size_t n) {
    size_t i = (size_t)blockIdx.x * blockDim.x + threadIdx.x;
    size_t stride = (size_t)gridDim.x * blockDim.x;
    for (; i < n; i += stride) p[i] = 0;
}

__global__ __launch_bounds__(256) void rmsnorm_kernel(const float* __restrict__ x,
                                                      const float* __restrict__ w,
                                                      bf16_t* __restrict__ out) {
    int row = blockIdx.x;
    const float* xr = x + (size_t)row * H_;
    __shared__ float red[256];
    float s = 0.f;
    for (int i = threadIdx.x; i < H_; i += 256) { float v = xr[i]; s += v * v; }
    red[threadIdx.x] = s;
    __syncthreads();
    for (int st = 128; st > 0; st >>= 1) {
        if (threadIdx.x < st) red[threadIdx.x] += red[threadIdx.x + st];
        __syncthreads();
    }
    float rs = rsqrtf(red[0] / (float)H_ + 1e-6f);
    for (int i = threadIdx.x; i < H_; i += 256)
        out[(size_t)row * H_ + i] = (bf16_t)(xr[i] * rs * w[i]);
}

// qlin: (B*T, NHeads*D) f32 -> out: (B, NHeads, T, D) bf16 (optionally RoPE'd)
__global__ void rope_kernel(const float* __restrict__ qlin,
                            const int* __restrict__ pos_ids,
                            bf16_t* __restrict__ out, int NHeads, int do_rope) {
    int idx = blockIdx.x * blockDim.x + threadIdx.x;
    int total = B_ * T_ * NHeads * (D_ / 2);
    if (idx >= total) return;
    int i = idx & 31;
    int tmp = idx >> 5;
    int h = tmp % NHeads; tmp /= NHeads;
    int t = tmp % T_;
    int b = tmp / T_;
    size_t base_in = ((size_t)(b * T_ + t) * NHeads + h) * D_;
    float x1 = qlin[base_in + i], x2 = qlin[base_in + i + 32];
    float o1 = x1, o2 = x2;
    if (do_rope) {
        float p = (float)pos_ids[b * T_ + t];
        float inv = __expf(-(2.f * (float)i / (float)D_) * __logf(1000000.f));
        float ang = p * inv;
        float c = __cosf(ang), s = __sinf(ang);
        o1 = x1 * c - x2 * s;
        o2 = x2 * c + x1 * s;
    }
    size_t base_out = ((size_t)(b * NHeads + h) * T_ + t) * D_;
    out[base_out + i] = (bf16_t)o1;
    out[base_out + i + 32] = (bf16_t)o2;
}

// ---------------- generic dense GEMM: C = A(MxK) * W(NxK)^T ----------------
// Block = 4 waves, block tile 16(M) x 256(N); each wave does 16x64 (4 wmma/A-frag).
// A tile is staged in LDS via the async-to-LDS engine and shared by all waves.
// MODE 0: store f32; 1: store f32 + residual; 2: silu -> bf16; 3: add into f32
template <int MODE>
__global__ __launch_bounds__(128) void gemm_bf16_kernel(
    const bf16_t* __restrict__ A, const bf16_t* __restrict__ W, int N, int K,
    const float* __restrict__ resid, float* __restrict__ outf,
    bf16_t* __restrict__ outb) {
    constexpr int KW = 1024;                 // K staging chunk (K % 1024 == 0)
    __shared__ bf16_t Alds[16 * KW];         // 32 KB
    int tid = threadIdx.x;
    int wave = tid >> 5, lane = tid & 31;
    int lo = lane & 15, hi = lane >> 4;
    int m0 = blockIdx.y * 16;
    int n0 = blockIdx.x * 256 + wave * 64;

    v8f acc[4] = {};
    for (int kc = 0; kc < K; kc += KW) {
        // cooperative async stage of A tile (16 rows x KW)
        for (int c = tid; c < 16 * KW / 8; c += 128) {
            int row = c / (KW / 8);
            int col = (c % (KW / 8)) * 8;
            async_copy16(A + (size_t)(m0 + row) * K + kc + col,
                         &Alds[row * KW + col]);
        }
        async_join();
        __syncthreads();
        for (int k0 = 0; k0 < KW; k0 += 32) {
            v16bf a = load_a_frag(Alds, KW, 0, k0, lane);
            __builtin_prefetch(W + (size_t)(n0 + lo) * K + kc + k0 + 256, 0, 1);
#pragma unroll
            for (int j = 0; j < 4; ++j) {
                v16bf b = load_a_frag(W, K, n0 + j * 16, kc + k0, lane);
                acc[j] = wmma_bf(a, b, acc[j]);
            }
        }
        __syncthreads();
    }
#pragma unroll
    for (int j = 0; j < 4; ++j) {
#pragma unroll
        for (int r = 0; r < 8; ++r) {
            size_t idx = (size_t)(m0 + hi * 8 + r) * N + n0 + j * 16 + lo;
            float c = acc[j][r];
            if (MODE == 0) outf[idx] = c;
            else if (MODE == 1) outf[idx] = resid[idx] + c;
            else if (MODE == 2) outb[idx] = (bf16_t)silu(c);
            else outf[idx] += c;
        }
    }
}

// ---------------- flash attention (1 wave / 16-query tile) ----------------
__global__ __launch_bounds__(32) void attn_kernel(const bf16_t* __restrict__ qb,
                                                  const bf16_t* __restrict__ kb,
                                                  const bf16_t* __restrict__ vb,
                                                  bf16_t* __restrict__ ob) {
    int lane = threadIdx.x, lo = lane & 15, hi = lane >> 4;
    int qt = blockIdx.x, bh = blockIdx.y;
    int b = bh / NH_, h = bh % NH_, kvh = h / (NH_ / NKV_);
    int q0 = qt * 16;
    const bf16_t* Q = qb + ((size_t)(b * NH_ + h) * T_) * D_;
    const bf16_t* Kp = kb + ((size_t)(b * NKV_ + kvh) * T_) * D_;
    const bf16_t* Vp = vb + ((size_t)(b * NKV_ + kvh) * T_) * D_;

    v16bf qa0 = load_a_frag(Q, D_, q0, 0, lane);
    v16bf qa1 = load_a_frag(Q, D_, q0, 32, lane);
    v8f o0 = {}, o1 = {}, o2 = {}, o3 = {};
    float mrow[8], lrow[8];
#pragma unroll
    for (int r = 0; r < 8; ++r) { mrow[r] = -3.0e38f; lrow[r] = 0.f; }

    __shared__ bf16_t Plds[16][32];
    __shared__ bf16_t Vlds[32 * 64];  // 4 KB V tile, async-staged

    for (int k0 = 0; k0 <= q0 + 15; k0 += 32) {
        // async-stage V tile (32 keys x 64 d) while S is computed
        for (int c = lane; c < 256; c += 32) {
            int row = c >> 3;
            int col = (c & 7) * 8;
            async_copy16(Vp + (size_t)(k0 + row) * D_ + col, &Vlds[row * 64 + col]);
        }
        // S = Q K^T for two 16-key subtiles (K rows are key-major, stride D)
        v8f s0 = {}, s1 = {};
        s0 = wmma_bf(qa0, load_a_frag(Kp, D_, k0, 0, lane), s0);
        s0 = wmma_bf(qa1, load_a_frag(Kp, D_, k0, 32, lane), s0);
        s1 = wmma_bf(qa0, load_a_frag(Kp, D_, k0 + 16, 0, lane), s1);
        s1 = wmma_bf(qa1, load_a_frag(Kp, D_, k0 + 16, 32, lane), s1);

        float alpha[8];
#pragma unroll
        for (int r = 0; r < 8; ++r) {
            int m = q0 + hi * 8 + r;
            s0[r] = (k0 + lo <= m) ? s0[r] * 0.125f : -3.0e38f;
            s1[r] = (k0 + 16 + lo <= m) ? s1[r] * 0.125f : -3.0e38f;
            float mx = fmaxf(s0[r], s1[r]);
#pragma unroll
            for (int off = 8; off >= 1; off >>= 1)
                mx = fmaxf(mx, __shfl_xor(mx, off, 32));
            float mnew = fmaxf(mrow[r], mx);
            alpha[r] = __expf(mrow[r] - mnew);
            float p0 = __expf(s0[r] - mnew);
            float p1 = __expf(s1[r] - mnew);
            s0[r] = p0; s1[r] = p1;
            float sum = p0 + p1;
#pragma unroll
            for (int off = 8; off >= 1; off >>= 1)
                sum += __shfl_xor(sum, off, 32);
            lrow[r] = alpha[r] * lrow[r] + sum;
            mrow[r] = mnew;
        }
#pragma unroll
        for (int r = 0; r < 8; ++r) {
            o0[r] *= alpha[r]; o1[r] *= alpha[r];
            o2[r] *= alpha[r]; o3[r] *= alpha[r];
        }
        // transpose P from C/D layout to A layout via LDS
#pragma unroll
        for (int r = 0; r < 8; ++r) {
            Plds[hi * 8 + r][lo] = (bf16_t)s0[r];
            Plds[hi * 8 + r][16 + lo] = (bf16_t)s1[r];
        }
        __syncthreads();
        v16bf pa = load_a_frag(&Plds[0][0], 32, 0, 0, lane);
        async_join();
        __syncthreads();
        o0 = wmma_bf(pa, load_b_strided(Vlds, 64, 0, 0, lane), o0);
        o1 = wmma_bf(pa, load_b_strided(Vlds, 64, 0, 16, lane), o1);
        o2 = wmma_bf(pa, load_b_strided(Vlds, 64, 0, 32, lane), o2);
        o3 = wmma_bf(pa, load_b_strided(Vlds, 64, 0, 48, lane), o3);
        __syncthreads();
    }
#pragma unroll
    for (int r = 0; r < 8; ++r) {
        int m = q0 + hi * 8 + r;
        float inv = 1.f / lrow[r];
        size_t base = (size_t)(b * T_ + m) * (NH_ * D_) + (size_t)h * D_;
        ob[base + lo] = (bf16_t)(o0[r] * inv);
        ob[base + 16 + lo] = (bf16_t)(o1[r] * inv);
        ob[base + 32 + lo] = (bf16_t)(o2[r] * inv);
        ob[base + 48 + lo] = (bf16_t)(o3[r] * inv);
    }
}

// ---------------- MoE routing ----------------
__global__ __launch_bounds__(256) void route_kernel(
    const bf16_t* __restrict__ h2, const float* __restrict__ gw,
    int* __restrict__ cnt, int* __restrict__ toklist, float* __restrict__ wtlist,
    float* __restrict__ imp, float* __restrict__ loadc) {
    int tok = blockIdx.x;
    __shared__ float sm[256][E_];
    float part[E_];
#pragma unroll
    for (int e = 0; e < E_; ++e) part[e] = 0.f;
    for (int hh = threadIdx.x; hh < H_; hh += 256) {
        float x = (float)h2[(size_t)tok * H_ + hh];
#pragma unroll
        for (int e = 0; e < E_; ++e) part[e] += x * gw[e * H_ + hh];
    }
#pragma unroll
    for (int e = 0; e < E_; ++e) sm[threadIdx.x][e] = part[e];
    __syncthreads();
    for (int st = 128; st > 0; st >>= 1) {
        if (threadIdx.x < st)
#pragma unroll
            for (int e = 0; e < E_; ++e) sm[threadIdx.x][e] += sm[threadIdx.x + st][e];
        __syncthreads();
    }
    if (threadIdx.x == 0) {
        float lg[E_], mx = -3.0e38f;
        for (int e = 0; e < E_; ++e) { lg[e] = sm[0][e]; mx = fmaxf(mx, lg[e]); }
        float den = 0.f;
        for (int e = 0; e < E_; ++e) { lg[e] = __expf(lg[e] - mx); den += lg[e]; }
        float pr[E_];
        for (int e = 0; e < E_; ++e) { pr[e] = lg[e] / den; atomicAdd(&imp[e], pr[e]); }
        int i1 = 0;
        for (int e = 1; e < E_; ++e) if (pr[e] > pr[i1]) i1 = e;
        int i2 = (i1 == 0) ? 1 : 0;
        for (int e = 0; e < E_; ++e) if (e != i1 && pr[e] > pr[i2]) i2 = e;
        int sel[2] = {i1, i2};
#pragma unroll
        for (int kk = 0; kk < KTOP_; ++kk) {
            int e = sel[kk];
            int p = atomicAdd(&cnt[e], 1);
            toklist[e * NT_ + p] = tok;
            wtlist[e * NT_ + p] = pr[e];
            atomicAdd(&loadc[e], 1.f);
        }
    }
}

// ---------------- MoE expert GEMMs (gathered, 16x64 per wave) ----------------
__global__ __launch_bounds__(32) void moe_w1_kernel(const bf16_t* __restrict__ h2,
                                                    const bf16_t* __restrict__ w1,
                                                    const int* __restrict__ cnt,
                                                    const int* __restrict__ toklist,
                                                    bf16_t* __restrict__ act) {
    int e = blockIdx.y, tile = blockIdx.x;
    int c = cnt[e];
    if (tile * 16 >= c) return;
    int lane = threadIdx.x, lo = lane & 15, hi = lane >> 4;
    int pos = tile * 16 + lo;
    int tokA = (pos < c) ? toklist[e * NT_ + pos] : 0;
    const bf16_t* Arow = h2 + (size_t)tokA * H_;
    const bf16_t* W = w1 + (size_t)e * I_ * H_;
    for (int n0 = 0; n0 < I_; n0 += 64) {
        v8f acc[4] = {};
        for (int k0 = 0; k0 < H_; k0 += 32) {
            v16bf a = load_a_row(Arow, k0, hi);
#pragma unroll
            for (int j = 0; j < 4; ++j)
                acc[j] = wmma_bf(a, load_a_frag(W, H_, n0 + j * 16, k0, lane), acc[j]);
        }
#pragma unroll
        for (int j = 0; j < 4; ++j)
#pragma unroll
            for (int r = 0; r < 8; ++r) {
                int rowidx = e * NT_ + tile * 16 + hi * 8 + r;
                act[(size_t)rowidx * I_ + n0 + j * 16 + lo] = (bf16_t)silu(acc[j][r]);
            }
    }
}

__global__ __launch_bounds__(32) void moe_w2_kernel(const bf16_t* __restrict__ act,
                                                    const bf16_t* __restrict__ w2,
                                                    const int* __restrict__ cnt,
                                                    const int* __restrict__ toklist,
                                                    const float* __restrict__ wtlist,
                                                    float* __restrict__ y) {
    int e = blockIdx.y, tile = blockIdx.x;
    int c = cnt[e];
    if (tile * 16 >= c) return;
    int lane = threadIdx.x, lo = lane & 15, hi = lane >> 4;
    int tokr[8]; float wtr[8];
#pragma unroll
    for (int r = 0; r < 8; ++r) {
        int pos = tile * 16 + hi * 8 + r;
        bool ok = pos < c;
        tokr[r] = ok ? toklist[e * NT_ + pos] : 0;
        wtr[r] = ok ? wtlist[e * NT_ + pos] : 0.f;
    }
    const bf16_t* A = act + (size_t)(e * NT_ + tile * 16) * I_;
    const bf16_t* W = w2 + (size_t)e * H_ * I_;
    for (int n0 = 0; n0 < H_; n0 += 64) {
        v8f acc[4] = {};
        for (int k0 = 0; k0 < I_; k0 += 32) {
            v16bf a = load_a_frag(A, I_, 0, k0, lane);
#pragma unroll
            for (int j = 0; j < 4; ++j)
                acc[j] = wmma_bf(a, load_a_frag(W, I_, n0 + j * 16, k0, lane), acc[j]);
        }
#pragma unroll
        for (int j = 0; j < 4; ++j)
#pragma unroll
            for (int r = 0; r < 8; ++r)
                atomicAdd(&y[(size_t)tokr[r] * H_ + n0 + j * 16 + lo],
                          wtr[r] * acc[j][r]);
    }
}

// ---------------- finalize ----------------
__global__ void final_add_kernel(const float* __restrict__ x1,
                                 const float* __restrict__ y,
                                 float* __restrict__ out) {
    size_t i = (size_t)blockIdx.x * blockDim.x + threadIdx.x;
    size_t n = (size_t)NT_ * H_;
    size_t stride = (size_t)gridDim.x * blockDim.x;
    for (; i < n; i += stride) out[i] = x1[i] + y[i];
}

__global__ void aux_kernel(const float* __restrict__ imp,
                           const float* __restrict__ loadc,
                           float* __restrict__ out_aux) {
    if (threadIdx.x == 0 && blockIdx.x == 0) {
        float s = 0.f;
        for (int e = 0; e < E_; ++e)
            s += (imp[e] / (float)NT_) * (loadc[e] / (float)(NT_ * KTOP_));
        *out_aux = s * (float)E_;
    }
}

// ---------------- host orchestration ----------------
extern "C" void kernel_launch(void* const* d_in, const int* in_sizes, int n_in,
                              void* d_out, int out_size, void* d_ws, size_t ws_size,
                              hipStream_t stream) {
    const float* x   = (const float*)d_in[0];
    const int*   pos = (const int*)d_in[1];
    const float* ln1 = (const float*)d_in[2];
    const float* ln2 = (const float*)d_in[3];
    const float* wq  = (const float*)d_in[4];
    const float* wk  = (const float*)d_in[5];
    const float* wv  = (const float*)d_in[6];
    const float* wo  = (const float*)d_in[7];
    const float* w1  = (const float*)d_in[8];
    const float* w2  = (const float*)d_in[9];
    const float* sw1 = (const float*)d_in[10];
    const float* sw2 = (const float*)d_in[11];
    const float* gw  = (const float*)d_in[12];
    float* out = (float*)d_out;

    char* wsp = (char*)d_ws;
    size_t off = 0;
    auto alloc = [&](size_t bytes) -> void* {
        void* p = wsp + off;
        off = (off + bytes + 255) & ~(size_t)255;
        return p;
    };
    bf16_t* wq_bf  = (bf16_t*)alloc((size_t)NH_ * D_ * H_ * 2);
    bf16_t* wk_bf  = (bf16_t*)alloc((size_t)NKV_ * D_ * H_ * 2);
    bf16_t* wv_bf  = (bf16_t*)alloc((size_t)NKV_ * D_ * H_ * 2);
    bf16_t* wo_bf  = (bf16_t*)alloc((size_t)H_ * NH_ * D_ * 2);
    bf16_t* w1_bf  = (bf16_t*)alloc((size_t)E_ * I_ * H_ * 2);
    bf16_t* w2_bf  = (bf16_t*)alloc((size_t)E_ * H_ * I_ * 2);
    bf16_t* sw1_bf = (bf16_t*)alloc((size_t)SI_ * H_ * 2);
    bf16_t* sw2_bf = (bf16_t*)alloc((size_t)H_ * SI_ * 2);
    bf16_t* h_bf   = (bf16_t*)alloc((size_t)NT_ * H_ * 2);
    float*  qlin   = (float*)alloc((size_t)NT_ * NH_ * D_ * 4);
    float*  klin   = (float*)alloc((size_t)NT_ * NKV_ * D_ * 4);
    float*  vlin   = (float*)alloc((size_t)NT_ * NKV_ * D_ * 4);
    bf16_t* q_bf   = (bf16_t*)alloc((size_t)NT_ * NH_ * D_ * 2);
    bf16_t* k_bf   = (bf16_t*)alloc((size_t)NT_ * NKV_ * D_ * 2);
    bf16_t* v_bf   = (bf16_t*)alloc((size_t)NT_ * NKV_ * D_ * 2);
    bf16_t* o_bf   = (bf16_t*)alloc((size_t)NT_ * NH_ * D_ * 2);
    float*  x1     = (float*)alloc((size_t)NT_ * H_ * 4);
    bf16_t* h2_bf  = (bf16_t*)alloc((size_t)NT_ * H_ * 2);
    bf16_t* act_m  = (bf16_t*)alloc((size_t)E_ * NT_ * I_ * 2);
    bf16_t* act_s  = (bf16_t*)alloc((size_t)NT_ * SI_ * 2);
    float*  yb     = (float*)alloc((size_t)NT_ * H_ * 4);
    int*    cnt    = (int*)alloc(E_ * 4);
    float*  imp    = (float*)alloc(E_ * 4);
    float*  loadc  = (float*)alloc(E_ * 4);
    int*    toklist = (int*)alloc((size_t)E_ * NT_ * 4);
    float*  wtlist  = (float*)alloc((size_t)E_ * NT_ * 4);

    // weights -> bf16
    f2bf_kernel<<<1024, 256, 0, stream>>>(wq, wq_bf, (size_t)NH_ * D_ * H_);
    f2bf_kernel<<<512, 256, 0, stream>>>(wk, wk_bf, (size_t)NKV_ * D_ * H_);
    f2bf_kernel<<<512, 256, 0, stream>>>(wv, wv_bf, (size_t)NKV_ * D_ * H_);
    f2bf_kernel<<<1024, 256, 0, stream>>>(wo, wo_bf, (size_t)H_ * NH_ * D_);
    f2bf_kernel<<<2048, 256, 0, stream>>>(w1, w1_bf, (size_t)E_ * I_ * H_);
    f2bf_kernel<<<2048, 256, 0, stream>>>(w2, w2_bf, (size_t)E_ * H_ * I_);
    f2bf_kernel<<<512, 256, 0, stream>>>(sw1, sw1_bf, (size_t)SI_ * H_);
    f2bf_kernel<<<512, 256, 0, stream>>>(sw2, sw2_bf, (size_t)H_ * SI_);

    // norm1 + QKV projections (WMMA)
    rmsnorm_kernel<<<NT_, 256, 0, stream>>>(x, ln1, h_bf);
    gemm_bf16_kernel<0><<<dim3(NH_ * D_ / 256, NT_ / 16), 128, 0, stream>>>(
        h_bf, wq_bf, NH_ * D_, H_, nullptr, qlin, nullptr);
    gemm_bf16_kernel<0><<<dim3(NKV_ * D_ / 256, NT_ / 16), 128, 0, stream>>>(
        h_bf, wk_bf, NKV_ * D_, H_, nullptr, klin, nullptr);
    gemm_bf16_kernel<0><<<dim3(NKV_ * D_ / 256, NT_ / 16), 128, 0, stream>>>(
        h_bf, wv_bf, NKV_ * D_, H_, nullptr, vlin, nullptr);

    // RoPE + relayout to (B, heads, T, D)
    rope_kernel<<<(B_ * T_ * NH_ * 32 + 255) / 256, 256, 0, stream>>>(qlin, pos, q_bf, NH_, 1);
    rope_kernel<<<(B_ * T_ * NKV_ * 32 + 255) / 256, 256, 0, stream>>>(klin, pos, k_bf, NKV_, 1);
    rope_kernel<<<(B_ * T_ * NKV_ * 32 + 255) / 256, 256, 0, stream>>>(vlin, pos, v_bf, NKV_, 0);

    // flash attention
    attn_kernel<<<dim3(T_ / 16, B_ * NH_), 32, 0, stream>>>(q_bf, k_bf, v_bf, o_bf);

    // output projection + residual -> x1
    gemm_bf16_kernel<1><<<dim3(H_ / 256, NT_ / 16), 128, 0, stream>>>(
        o_bf, wo_bf, H_, NH_ * D_, x, x1, nullptr);

    // norm2
    rmsnorm_kernel<<<NT_, 256, 0, stream>>>(x1, ln2, h2_bf);

    // zero accumulators
    zero_f_kernel<<<2048, 256, 0, stream>>>(yb, (size_t)NT_ * H_);
    zero_f_kernel<<<1, 64, 0, stream>>>(imp, E_);
    zero_f_kernel<<<1, 64, 0, stream>>>(loadc, E_);
    zero_i_kernel<<<1, 64, 0, stream>>>(cnt, E_);

    // routing
    route_kernel<<<NT_, 256, 0, stream>>>(h2_bf, gw, cnt, toklist, wtlist, imp, loadc);

    // routed experts (WMMA gathered GEMMs)
    moe_w1_kernel<<<dim3(NT_ / 16, E_), 32, 0, stream>>>(h2_bf, w1_bf, cnt, toklist, act_m);
    moe_w2_kernel<<<dim3(NT_ / 16, E_), 32, 0, stream>>>(act_m, w2_bf, cnt, toklist, wtlist, yb);

    // shared expert
    gemm_bf16_kernel<2><<<dim3(SI_ / 256, NT_ / 16), 128, 0, stream>>>(
        h2_bf, sw1_bf, SI_, H_, nullptr, nullptr, act_s);
    gemm_bf16_kernel<3><<<dim3(H_ / 256, NT_ / 16), 128, 0, stream>>>(
        act_s, sw2_bf, H_, SI_, nullptr, yb, nullptr);

    // out = x1 + y ; aux loss
    final_add_kernel<<<4096, 256, 0, stream>>>(x1, yb, out);
    aux_kernel<<<1, 1, 0, stream>>>(imp, loadc, out + (size_t)NT_ * H_);
}
